// MoE_5935644803777
// MI455X (gfx1250) — matrix operand order repaired and checked by reading, not compiled, not run on hardware
//
#include <hip/hip_runtime.h>
#include <hip/hip_bf16.h>
#include <math.h>

// ---- problem constants (from reference) ----
#define N_TOK  2048   // B*S
#define DMODEL 1024   // D
#define GHID   256    // G
#define NEXP   8      // E
#define HEXP   2048   // H
#define TOPK   2      // K

typedef __attribute__((ext_vector_type(16))) __bf16 v16bf;
typedef __attribute__((ext_vector_type(8)))  __bf16 v8bf;
typedef __attribute__((ext_vector_type(8)))  float  v8f;
typedef __attribute__((ext_vector_type(4)))  unsigned v4u;
typedef __attribute__((ext_vector_type(8)))  unsigned v8u;

__device__ __forceinline__ float gelu_exact(float v) {
    return 0.5f * v * (1.0f + erff(v * 0.70710678118654752f));
}

// generic LDS pointer: low 32 bits are the LDS byte offset (ISA aperture rule)
__device__ __forceinline__ unsigned lds_off(const void* p) {
    return (unsigned)(unsigned long long)p;
}

// async global -> LDS, 16B per lane, tracked by ASYNCcnt
__device__ __forceinline__ void async_ld16(unsigned lds, const void* g) {
    asm volatile("global_load_async_to_lds_b128 %0, %1, off"
                 :: "v"(lds), "v"((unsigned long long)g) : "memory");
}
__device__ __forceinline__ void wait_async3() {
    asm volatile("s_wait_asynccnt 0x3" ::: "memory");
}
__device__ __forceinline__ void wait_async0() {
    asm volatile("s_wait_asynccnt 0x0" ::: "memory");
}

// ---- Tensor Data Mover: 2-D tile load (bf16 elems), LDS rows padded 64B+16B ----
// tile_d0 elems per row (must make 64B here: 32), tile_d1 rows, row stride in elems.
// pad_interval code 3 (16 DWORDS = 64B), pad_amount code 3 (4 DWORDS = 16B)
// -> LDS row pitch 80B, matching __bf16[rows][40].
__device__ __forceinline__ void tdm_load_2d(unsigned lds, unsigned long long gaddr,
                                            unsigned tile_d0, unsigned tile_d1,
                                            unsigned row_stride_elems) {
    v4u g0;
    g0[0] = 1u;                                   // count=1, user descriptor
    g0[1] = lds;                                  // lds_addr
    g0[2] = (unsigned)gaddr;                      // global_addr[31:0]
    g0[3] = (unsigned)(gaddr >> 32) | (2u << 30); // global_addr[56:32] | type=2
    v8u g1;
    g1[0] = 0x00010000u       // data_size=1 (2 bytes)
          | (1u << 20)        // pad_enable
          | (3u << 22)        // pad_interval: 16 DWORDs
          | (3u << 25);       // pad_amount: 4 DWORDs
    g1[1] = 0u;                        // atomic_barrier=0, tensor_dim0[15:0]=0
    g1[2] = 0x10u;                     // tensor_dim0=0x100000 (huge), tensor_dim1[15:0]=0
    g1[3] = (tile_d0 << 16) | 0x10u;   // tile_dim0 | tensor_dim1=0x100000
    g1[4] = tile_d1;                   // tile_dim1, tile_dim2=0
    g1[5] = row_stride_elems;          // tensor_dim0_stride[31:0]
    g1[6] = 0u;
    g1[7] = 0u;
    asm volatile("tensor_load_to_lds %0, %1" :: "s"(g0), "s"(g1) : "memory");
}

// ---------------- meta init ----------------
// meta ints: [0..7]=counts, [8..15]=cursors, [16..24]=offsets, [32..63]=zero pad (async src)
__global__ void k_init_meta(int* meta) {
    if (threadIdx.x < 64) meta[threadIdx.x] = 0;
}

// ---------------- x f32 -> bf16 ----------------
__global__ void k_cvt_x(const float* __restrict__ x, __bf16* __restrict__ xb, int n) {
    int i = blockIdx.x * 256 + threadIdx.x;
    if (i < n) xb[i] = (__bf16)x[i];
}

// ------------- tiled transpose + convert: in[R][C] f32 -> out[C][R] bf16 -------------
__global__ __launch_bounds__(256) void k_transpose_cvt(const float* __restrict__ in,
                                                       __bf16* __restrict__ out,
                                                       int R, int C) {
    __shared__ float tile[32][33];
    const float* src = in  + (size_t)blockIdx.z * R * C;
    __bf16*      dst = out + (size_t)blockIdx.z * R * C;
    int c0 = blockIdx.x * 32, r0 = blockIdx.y * 32;
    int tx = threadIdx.x, ty = threadIdx.y;
    #pragma unroll
    for (int i = 0; i < 4; ++i) {
        int r = r0 + ty + 8 * i;
        tile[ty + 8 * i][tx] = src[(size_t)r * C + c0 + tx];
    }
    __syncthreads();
    #pragma unroll
    for (int i = 0; i < 4; ++i) {
        int c = c0 + ty + 8 * i;
        dst[(size_t)c * R + r0 + tx] = (__bf16)tile[tx][ty + 8 * i];
    }
}

// ---------------- gating: fp32 MLP + softmax + top-2 ----------------
__global__ __launch_bounds__(256) void k_gate(const float* __restrict__ x,
                                              const float* __restrict__ wg1,
                                              const float* __restrict__ bg1,
                                              const float* __restrict__ wg2,
                                              int* __restrict__ meta,
                                              int* __restrict__ top_idx,
                                              float* __restrict__ top_val) {
    __shared__ float xs[DMODEL];
    __shared__ float gs[GHID];
    __shared__ float ls[NEXP];
    const int t = blockIdx.x;
    const int j = threadIdx.x;
    const float* xr = x + (size_t)t * DMODEL;
    for (int i = j; i < DMODEL; i += 256) xs[i] = xr[i];
    __syncthreads();

    float acc = bg1[j];
    #pragma unroll 4
    for (int d = 0; d < DMODEL; ++d)
        acc = fmaf(xs[d], wg1[(size_t)d * GHID + j], acc);
    gs[j] = gelu_exact(acc);
    __syncthreads();

    if (j < NEXP) {
        float s = 0.f;
        for (int i = 0; i < GHID; ++i)
            s = fmaf(gs[i], wg2[(size_t)i * NEXP + j], s);
        ls[j] = s;
    }
    __syncthreads();

    if (j == 0) {
        float mx = ls[0];
        for (int e = 1; e < NEXP; ++e) mx = fmaxf(mx, ls[e]);
        float p[NEXP], se = 0.f;
        for (int e = 0; e < NEXP; ++e) { p[e] = __expf(ls[e] - mx); se += p[e]; }
        float inv = 1.f / se;
        for (int e = 0; e < NEXP; ++e) p[e] *= inv;
        int i0 = 0;
        for (int e = 1; e < NEXP; ++e) if (p[e] > p[i0]) i0 = e;
        int i1 = -1;
        for (int e = 0; e < NEXP; ++e) {
            if (e == i0) continue;
            if (i1 < 0 || p[e] > p[i1]) i1 = e;
        }
        top_idx[t * 2]     = i0;  top_val[t * 2]     = p[i0];
        top_idx[t * 2 + 1] = i1;  top_val[t * 2 + 1] = p[i1];
        atomicAdd(&meta[i0], 1);
        atomicAdd(&meta[i1], 1);
    }
}

// ---------------- exclusive scan of per-expert counts ----------------
__global__ void k_scan(int* meta) {
    if (threadIdx.x == 0) {
        int run = 0;
        for (int e = 0; e < NEXP; ++e) { meta[16 + e] = run; run += meta[e]; }
        meta[16 + NEXP] = run;
    }
}

// ---------------- scatter tokens into expert-grouped slot lists ----------------
__global__ void k_scatter(const int* __restrict__ top_idx, int* __restrict__ meta,
                          int* __restrict__ token_ids, int* __restrict__ slot_of) {
    int t = blockIdx.x * 256 + threadIdx.x;
    if (t >= N_TOK) return;
    for (int k = 0; k < TOPK; ++k) {
        int e = top_idx[t * 2 + k];
        int pos = meta[16 + e] + atomicAdd(&meta[8 + e], 1);
        token_ids[pos]     = t;
        slot_of[t * 2 + k] = pos;
    }
}

// ---------------- WMMA fragment loaders (per CDNA5 ISA layouts) ----------------
__device__ __forceinline__ v16bf frag_a(const __bf16* rowp, int half) {
    v8bf lo = *(const v8bf*)(rowp + half * 8);
    v8bf hi = *(const v8bf*)(rowp + 16 + half * 8);
    v16bf a;
    #pragma unroll
    for (int i = 0; i < 8; ++i) { a[i] = lo[i]; a[8 + i] = hi[i]; }
    return a;
}
__device__ __forceinline__ v16bf frag_b(const __bf16* colp, int half) {
    v8bf lo = *(const v8bf*)(colp + half * 8);
    v8bf hi = *(const v8bf*)(colp + 16 + half * 8);
    v16bf b;
    #pragma unroll
    for (int i = 0; i < 8; ++i) { b[i] = lo[i]; b[8 + i] = hi[i]; }
    return b;
}

// ---------------- expert GEMM1: h = gelu(x_gathered @ w_in[e]) ----------------
// async global->LDS feeds + 2-deep pipeline; 8 waves: 2x4, 32x32 per wave
__global__ __launch_bounds__(256) void k_expert_in(const __bf16* __restrict__ xb,
                                                   const __bf16* __restrict__ w_in_t, // [E][H][D]
                                                   const int* __restrict__ meta,
                                                   const int* __restrict__ token_ids,
                                                   __bf16* __restrict__ h) {
    __shared__ __align__(16) __bf16 Asm[2][64][40];
    __shared__ __align__(16) __bf16 Bsm[2][128][40];
    const int e   = blockIdx.z;
    const int off = meta[16 + e];
    const int cnt = meta[16 + e + 1] - off;
    const int m0  = blockIdx.y * 64;
    if (m0 >= cnt) return;
    const int rem = cnt - m0;
    const int n0  = blockIdx.x * 128;

    const int tid  = threadIdx.x;
    const int lane = tid & 31;
    const int wv   = tid >> 5;
    const int wm   = (wv & 1) * 32;
    const int wn   = (wv >> 1) * 32;
    const int half = lane >> 4;
    const int ln   = lane & 15;

    // cooperative tile mapping: A 64x32 (8 bf16/thr), B 128x32 (16 bf16/thr)
    const int ar = tid >> 2, ac = (tid & 3) * 8;
    const int br = tid >> 1, bc = (tid & 1) * 16;
    const __bf16* zpad = (const __bf16*)(meta + 32);   // 64B of zeros
    const __bf16* arow = (ar < rem) ? (xb + (size_t)token_ids[off + m0 + ar] * DMODEL) : nullptr;
    const __bf16* brow = w_in_t + ((size_t)e * HEXP + n0 + br) * DMODEL;

    unsigned aL[2], bL0[2], bL1[2];
    #pragma unroll
    for (int s = 0; s < 2; ++s) {
        aL[s]  = lds_off(&Asm[s][ar][ac]);
        bL0[s] = lds_off(&Bsm[s][br][bc]);
        bL1[s] = lds_off(&Bsm[s][br][bc + 8]);
    }

    v8f c00 = {}, c01 = {}, c10 = {}, c11 = {};

    auto issue = [&](int s, int k0) {
        const __bf16* ga = arow ? (arow + k0 + ac) : zpad;
        async_ld16(aL[s],  ga);
        async_ld16(bL0[s], brow + k0 + bc);
        async_ld16(bL1[s], brow + k0 + bc + 8);
    };
    auto compute = [&](int s) {
        v16bf a0 = frag_a(&Asm[s][wm + ln][0], half);
        v16bf a1 = frag_a(&Asm[s][wm + 16 + ln][0], half);
        v16bf b0 = frag_b(&Bsm[s][wn + ln][0], half);
        v16bf b1 = frag_b(&Bsm[s][wn + 16 + ln][0], half);
        c00 = __builtin_amdgcn_wmma_f32_16x16x32_bf16(false, a0, false, b0, (short)0, c00, false, false);
        c01 = __builtin_amdgcn_wmma_f32_16x16x32_bf16(false, a0, false, b1, (short)0, c01, false, false);
        c10 = __builtin_amdgcn_wmma_f32_16x16x32_bf16(false, a1, false, b0, (short)0, c10, false, false);
        c11 = __builtin_amdgcn_wmma_f32_16x16x32_bf16(false, a1, false, b1, (short)0, c11, false, false);
    };

    issue(0, 0);
    int buf = 0;
    for (int k0 = 32; k0 < DMODEL; k0 += 32) {
        issue(buf ^ 1, k0);
        wait_async3();          // stage `buf` copies done (3 newest still in flight)
        __syncthreads();
        compute(buf);
        __syncthreads();        // all waves done reading `buf` before it is refilled
        buf ^= 1;
    }
    wait_async0();
    __syncthreads();
    compute(buf);

    // fused GELU + bf16 store. C layout: VGPR r -> M = r + 8*half, N = lane&15.
    #pragma unroll
    for (int mi = 0; mi < 2; ++mi) {
        v8f cc0 = mi ? c10 : c00;
        v8f cc1 = mi ? c11 : c01;
        #pragma unroll
        for (int r = 0; r < 8; ++r) {
            int lrow = wm + mi * 16 + r + 8 * half;
            if (lrow < rem) {
                __bf16* hp = h + (size_t)(off + m0 + lrow) * HEXP + n0;
                hp[wn + ln]      = (__bf16)gelu_exact(cc0[r]);
                hp[wn + 16 + ln] = (__bf16)gelu_exact(cc1[r]);
            }
        }
    }
}

// ---------------- expert GEMM2: o_part = h @ w_out[e] (TDM-fed tiles) ----------------
__global__ __launch_bounds__(256) void k_expert_out(const __bf16* __restrict__ h,
                                                    const __bf16* __restrict__ w_out_t, // [E][D][H]
                                                    const int* __restrict__ meta,
                                                    float* __restrict__ o_part) {
    __shared__ __align__(16) __bf16 Asm[2][64][40];
    __shared__ __align__(16) __bf16 Bsm[2][128][40];
    const int e   = blockIdx.z;
    const int off = meta[16 + e];
    const int cnt = meta[16 + e + 1] - off;
    const int m0  = blockIdx.y * 64;
    if (m0 >= cnt) return;
    const int rem = cnt - m0;
    const int n0  = blockIdx.x * 128;

    const int tid  = threadIdx.x;
    const int lane = tid & 31;
    const int wv   = tid >> 5;
    const int wm   = (wv & 1) * 32;
    const int wn   = (wv >> 1) * 32;
    const int half = lane >> 4;
    const int ln   = lane & 15;

    // h rows for this expert are contiguous; both tiles are plain 2-D strided -> TDM
    const unsigned long long gA = (unsigned long long)(h       + (size_t)(off + m0) * HEXP);
    const unsigned long long gB = (unsigned long long)(w_out_t + ((size_t)e * DMODEL + n0) * HEXP);
    const unsigned ldsA[2] = { lds_off(&Asm[0][0][0]), lds_off(&Asm[1][0][0]) };
    const unsigned ldsB[2] = { lds_off(&Bsm[0][0][0]), lds_off(&Bsm[1][0][0]) };
    const bool issuer = (tid < 32);   // one wave drives the TDM (EXEC ignored by tensor ops)

    v8f c00 = {}, c01 = {}, c10 = {}, c11 = {};

    auto issue = [&](int s, int k0) {
        tdm_load_2d(ldsA[s], gA + 2ull * k0, 32u, 64u,  (unsigned)HEXP);
        tdm_load_2d(ldsB[s], gB + 2ull * k0, 32u, 128u, (unsigned)HEXP);
    };
    auto compute = [&](int s) {
        v16bf a0 = frag_a(&Asm[s][wm + ln][0], half);
        v16bf a1 = frag_a(&Asm[s][wm + 16 + ln][0], half);
        v16bf b0 = frag_b(&Bsm[s][wn + ln][0], half);
        v16bf b1 = frag_b(&Bsm[s][wn + 16 + ln][0], half);
        c00 = __builtin_amdgcn_wmma_f32_16x16x32_bf16(false, a0, false, b0, (short)0, c00, false, false);
        c01 = __builtin_amdgcn_wmma_f32_16x16x32_bf16(false, a0, false, b1, (short)0, c01, false, false);
        c10 = __builtin_amdgcn_wmma_f32_16x16x32_bf16(false, a1, false, b0, (short)0, c10, false, false);
        c11 = __builtin_amdgcn_wmma_f32_16x16x32_bf16(false, a1, false, b1, (short)0, c11, false, false);
    };

    if (issuer) issue(0, 0);
    int buf = 0;
    for (int k0 = 32; k0 < HEXP; k0 += 32) {
        if (issuer) {
            issue(buf ^ 1, k0);
            __builtin_amdgcn_s_wait_tensorcnt(2);   // stage `buf` TDM ops complete
        }
        __syncthreads();
        compute(buf);
        __syncthreads();
        buf ^= 1;
    }
    if (issuer) __builtin_amdgcn_s_wait_tensorcnt(0);
    __syncthreads();
    compute(buf);

    #pragma unroll
    for (int mi = 0; mi < 2; ++mi) {
        v8f cc0 = mi ? c10 : c00;
        v8f cc1 = mi ? c11 : c01;
        #pragma unroll
        for (int r = 0; r < 8; ++r) {
            int lrow = wm + mi * 16 + r + 8 * half;
            if (lrow < rem) {
                float* op = o_part + (size_t)(off + m0 + lrow) * DMODEL + n0;
                op[wn + ln]      = cc0[r];
                op[wn + 16 + ln] = cc1[r];
            }
        }
    }
}

// ---------------- combine: y[t] = g0*o[slot0] + g1*o[slot1] (deterministic) ----------------
__global__ void k_combine(const float* __restrict__ o_part, const int* __restrict__ slot_of,
                          const float* __restrict__ top_val, float* __restrict__ y, int total) {
    int i = blockIdx.x * 256 + threadIdx.x;
    if (i >= total) return;
    int t = i / DMODEL, c = i % DMODEL;
    int s0 = slot_of[t * 2], s1 = slot_of[t * 2 + 1];
    y[i] = top_val[t * 2]     * o_part[(size_t)s0 * DMODEL + c] +
           top_val[t * 2 + 1] * o_part[(size_t)s1 * DMODEL + c];
}

// ---------------- host launcher ----------------
extern "C" void kernel_launch(void* const* d_in, const int* in_sizes, int n_in,
                              void* d_out, int out_size, void* d_ws, size_t ws_size,
                              hipStream_t stream) {
    const float* x     = (const float*)d_in[0];
    const float* wg1   = (const float*)d_in[1];
    const float* bg1   = (const float*)d_in[2];
    const float* wg2   = (const float*)d_in[3];
    const float* w_in  = (const float*)d_in[4];
    const float* w_out = (const float*)d_in[5];
    float* y = (float*)d_out;

    char* ws = (char*)d_ws;
    size_t off = 0;
    auto carve = [&](size_t bytes) -> char* {
        char* p = ws + off;
        off = (off + bytes + 255) & ~(size_t)255;
        return p;
    };
    int*    meta      = (int*)   carve(256);
    int*    top_idx   = (int*)   carve(sizeof(int)   * N_TOK * TOPK);
    float*  top_val   = (float*) carve(sizeof(float) * N_TOK * TOPK);
    int*    token_ids = (int*)   carve(sizeof(int)   * N_TOK * TOPK);
    int*    slot_of   = (int*)   carve(sizeof(int)   * N_TOK * TOPK);
    __bf16* xb        = (__bf16*)carve(sizeof(__bf16) * (size_t)N_TOK * DMODEL);
    __bf16* w_in_t    = (__bf16*)carve(sizeof(__bf16) * (size_t)NEXP * DMODEL * HEXP);
    __bf16* w_out_t   = (__bf16*)carve(sizeof(__bf16) * (size_t)NEXP * DMODEL * HEXP);
    __bf16* hbuf      = (__bf16*)carve(sizeof(__bf16) * (size_t)N_TOK * TOPK * HEXP);
    float*  o_part    = (float*) carve(sizeof(float)  * (size_t)N_TOK * TOPK * DMODEL);

    k_init_meta<<<1, 64, 0, stream>>>(meta);
    k_cvt_x<<<(N_TOK * DMODEL + 255) / 256, 256, 0, stream>>>(x, xb, N_TOK * DMODEL);
    // w_in [E][D][H] -> w_in_t [E][H][D] ; w_out [E][H][D] -> w_out_t [E][D][H]
    k_transpose_cvt<<<dim3(HEXP / 32, DMODEL / 32, NEXP), dim3(32, 8), 0, stream>>>(w_in, w_in_t, DMODEL, HEXP);
    k_transpose_cvt<<<dim3(DMODEL / 32, HEXP / 32, NEXP), dim3(32, 8), 0, stream>>>(w_out, w_out_t, HEXP, DMODEL);
    k_gate<<<N_TOK, 256, 0, stream>>>(x, wg1, bg1, wg2, meta, top_idx, top_val);
    k_scan<<<1, 32, 0, stream>>>(meta);
    k_scatter<<<N_TOK / 256, 256, 0, stream>>>(top_idx, meta, token_ids, slot_of);
    k_expert_in<<<dim3(HEXP / 128, N_TOK / 64, NEXP), 256, 0, stream>>>(xb, w_in_t, meta, token_ids, hbuf);
    k_expert_out<<<dim3(DMODEL / 128, N_TOK / 64, NEXP), 256, 0, stream>>>(hbuf, w_out_t, meta, o_part);
    k_combine<<<(N_TOK * DMODEL + 255) / 256, 256, 0, stream>>>(o_part, slot_of, top_val, y, N_TOK * DMODEL);
}
